// TinyViTBlock_13245679140954
// MI455X (gfx1250) — compile-verified, matrix operand order checked by reading, not run on hardware
//
#include <hip/hip_runtime.h>
#include <hip/hip_bf16.h>

typedef __attribute__((ext_vector_type(16))) _Float16 v16h;
typedef __attribute__((ext_vector_type(8)))  float    v8f;
typedef __attribute__((ext_vector_type(4)))  unsigned int v4u;
typedef __attribute__((ext_vector_type(4)))  int      v4i;
typedef __attribute__((ext_vector_type(8)))  int      v8i;
typedef _Float16 half_t;

#define H_     64
#define W_     64
#define HW_    (H_*W_)
#define DIM_   384
#define NH_    12
#define HD_    32
#define WS_    7
#define NTOK   49          // tokens per window
#define NWIN   400         // 4 * 10 * 10
#define MW     19600       // NWIN*NTOK
#define MWP    19712       // padded to 128
#define BATCH_ 4
#define MX     16384       // BATCH_*HW_
#define HID_   1536
#define QKVN   1152
#define SCALE_ 0.17677669529663687f
#define EPS_   1e-5f

#if __has_builtin(__builtin_amdgcn_tensor_load_to_lds)
#define HAVE_TDM 1
#else
#define HAVE_TDM 0
#endif

__device__ __forceinline__ int iabs_(int a){ return a < 0 ? -a : a; }

__device__ __forceinline__ v8f zero8(){
    v8f z;
#pragma unroll
    for (int i = 0; i < 8; ++i) z[i] = 0.0f;
    return z;
}

// Load a 16x32 f16 A/B-operand fragment from LDS (row-major, `stride` halves).
// CDNA5 16-bit A layout: lane L holds row (L&15); lanes 0-15 hold K 0..7,16..23,
// lanes 16-31 hold K 8..15,24..31 -> two b128 loads per lane.
__device__ __forceinline__ v16h load_frag(const half_t* base, int stride, int row, int lane, int kbase){
    union { v16h h; uint4 u[2]; } f;
    const int kh = (lane >> 4) * 8;
    const half_t* p = base + row * stride + kbase + kh;
    f.u[0] = *reinterpret_cast<const uint4*>(p);
    f.u[1] = *reinterpret_cast<const uint4*>(p + 16);
    return f.h;
}

#if HAVE_TDM
// Issue one TDM 2D tile load: 32 halves x 128 rows, row stride Kelems halves,
// destination LDS gets 16B padding after every 64B row (row stride 80B = 40 halves).
// D# per CDNA5 ISA ch.8: group0 {count=1, lds_addr, global_addr, type=2},
// group1 {data_size=2B, pad_enable, pad_interval=16dw, pad_amount=4dw,
//         tensor_dim0=32, tensor_dim1=128, tile 32x128, dim0_stride=Kelems}.
__device__ __forceinline__ void tdm_tile_load(const half_t* gsrc, unsigned lds_addr, int Kelems)
{
    unsigned long long ga = (unsigned long long)(uintptr_t)gsrc;
    v4u g0;
    g0.x = 1u;                                   // count=1 (valid descriptor)
    g0.y = lds_addr;                             // LDS byte address
    g0.z = (unsigned)ga;                         // global_addr[31:0]
    g0.w = ((unsigned)(ga >> 32) & 0x01FFFFFFu)  // global_addr[56:32]
         | 0x80000000u;                          // type=2 ("image")
    v8i g1;
    g1[0] = (int)((1u<<16) | (1u<<20) | (3u<<22) | (3u<<25)); // 2B, pad_en, 16dw interval, 4dw pad
    g1[1] = (int)(32u  << 16);   // tensor_dim0 = 32
    g1[2] = (int)(128u << 16);   // tensor_dim1(lo16)=128 ; tensor_dim0(hi16)=0
    g1[3] = (int)(32u  << 16);   // tile_dim0 = 32 ; tensor_dim1(hi16)=0
    g1[4] = 128;                 // tile_dim1 = 128 ; tile_dim2 = 0
    g1[5] = Kelems;              // tensor_dim0_stride[31:0]
    g1[6] = 0;
    g1[7] = 0;
    v4i z4 = {0,0,0,0};
#if defined(__clang_major__) && (__clang_major__ >= 23)
    v8i z8 = {0,0,0,0,0,0,0,0};
    __builtin_amdgcn_tensor_load_to_lds(g0, g1, z4, z4, z8, 0);
#else
    __builtin_amdgcn_tensor_load_to_lds(g0, g1, z4, z4, 0);
#endif
}
#endif

// ---------------------------------------------------------------------------
// LayerNorm -> f16. windowed=1: rows are window-major MWP tokens of x (with
// 70x70 zero pad); windowed=0: rows are direct (b,hw) tokens of src.
// ---------------------------------------------------------------------------
__global__ __launch_bounds__(128)
void ln_kernel(const float* __restrict__ src, const float* __restrict__ g,
               const float* __restrict__ b, half_t* __restrict__ out, int windowed)
{
    __shared__ float red[128];
    const int m   = blockIdx.x;
    const int tid = threadIdx.x;

    size_t sbase = 0;
    bool   valid = true;
    if (windowed) {
        if (m >= MW) { // zero M-pad rows
#pragma unroll
            for (int i = 0; i < 3; ++i) out[(size_t)m*DIM_ + tid + i*128] = (half_t)0.0f;
            return;
        }
        int win = m / NTOK, n = m % NTOK;
        int bb = win / 100, wr = (win % 100) / 10, wc = win % 10;
        int y = wr*WS_ + n/WS_, x = wc*WS_ + n%WS_;
        valid = (y < H_) && (x < W_);
        sbase = ((size_t)bb*HW_ + (size_t)y*W_ + x) * DIM_;
    } else {
        sbase = (size_t)m * DIM_;
    }

    float v[3], s = 0.f, ss = 0.f;
#pragma unroll
    for (int i = 0; i < 3; ++i) {
        float t = valid ? src[sbase + tid + i*128] : 0.0f;
        v[i] = t; s += t; ss += t*t;
    }
    red[tid] = s; __syncthreads();
    for (int off = 64; off > 0; off >>= 1) { if (tid < off) red[tid] += red[tid+off]; __syncthreads(); }
    float mean = red[0] / (float)DIM_; __syncthreads();
    red[tid] = ss; __syncthreads();
    for (int off = 64; off > 0; off >>= 1) { if (tid < off) red[tid] += red[tid+off]; __syncthreads(); }
    float var  = red[0] / (float)DIM_ - mean*mean;
    float rstd = rsqrtf(var + EPS_);
#pragma unroll
    for (int i = 0; i < 3; ++i) {
        int c = tid + i*128;
        out[(size_t)m*DIM_ + c] = (half_t)((v[i]-mean)*rstd*g[c] + b[c]);
    }
}

// ---------------------------------------------------------------------------
// Weight convert+transpose: src[K][N] f32 -> dst[N][K] f16 (coalesced writes).
// ---------------------------------------------------------------------------
__global__ void transpose_cvt(const float* __restrict__ src, half_t* __restrict__ dst,
                              int Krows, int Ncols)
{
    int idx = blockIdx.x * 256 + threadIdx.x;
    if (idx >= Krows * Ncols) return;
    int n = idx / Krows, k = idx % Krows;
    dst[idx] = (half_t)src[(size_t)k * Ncols + n];
}

// ---------------------------------------------------------------------------
// WMMA GEMM: C[M,N] = A[M,K](f16) * Bt[N,K](f16)^T, f32 accumulate.
// Block tile 128x128, BK=32, 8 waves of 32x64, LDS ping-pong.
// Tiles reach LDS via the Tensor Data Mover (pad_amount swizzles rows to the
// conflict-free 40-half stride); falls back to VGPR staging without TDM.
// MODE 0: out f16 = acc + bias                     (QKV)
// MODE 1: out f32 = resid + acc + bias, un-window  (proj + residual)
// MODE 2: out f16 = gelu(acc + bias)               (fc1)
// MODE 3: out f32 = acc + bias + resid             (fc2 + residual)
// ---------------------------------------------------------------------------
template<int MODE>
__global__ __launch_bounds__(256)
void gemm_wmma(const half_t* __restrict__ A, const half_t* __restrict__ Bt,
               const float* __restrict__ bias, void* __restrict__ out,
               const float* __restrict__ resid, int M, int N, int K)
{
    constexpr int BK = 32, LSTR = 40;
    __shared__ __align__(16) half_t As[2][128*LSTR];
    __shared__ __align__(16) half_t Bs[2][128*LSTR];

    const int tid  = threadIdx.x;
    const int lane = tid & 31;
    const int wave = tid >> 5;
    const int wm   = wave & 3;      // 4 waves along M
    const int wn   = wave >> 2;     // 2 waves along N
    const int bm   = blockIdx.y * 128;
    const int bn   = blockIdx.x * 128;
    const int KT   = K / BK;

    v8f acc[2][4];
#pragma unroll
    for (int i = 0; i < 2; ++i)
#pragma unroll
        for (int j = 0; j < 4; ++j) acc[i][j] = zero8();

#if HAVE_TDM
    auto issue_tiles = [&](int kt, int buf){
        tdm_tile_load(A  + (size_t)bm*K + kt*BK, (unsigned)(uintptr_t)&As[buf][0], K);
        tdm_tile_load(Bt + (size_t)bn*K + kt*BK, (unsigned)(uintptr_t)&Bs[buf][0], K);
    };
    if (wave == 0) issue_tiles(0, 0);
    for (int kt = 0; kt < KT; ++kt) {
        if (wave == 0) __builtin_amdgcn_s_wait_tensorcnt(0);
        __syncthreads();
        if (wave == 0 && kt + 1 < KT) issue_tiles(kt + 1, (kt + 1) & 1);
        const int buf = kt & 1;
        v16h af[2], bf[4];
#pragma unroll
        for (int mt = 0; mt < 2; ++mt)
            af[mt] = load_frag(&As[buf][0], LSTR, wm*32 + mt*16 + (lane & 15), lane, 0);
#pragma unroll
        for (int nt = 0; nt < 4; ++nt)
            bf[nt] = load_frag(&Bs[buf][0], LSTR, wn*64 + nt*16 + (lane & 15), lane, 0);
#pragma unroll
        for (int mt = 0; mt < 2; ++mt)
#pragma unroll
            for (int nt = 0; nt < 4; ++nt)
                acc[mt][nt] = __builtin_amdgcn_wmma_f32_16x16x32_f16(
                    false, af[mt], false, bf[nt], (short)0, acc[mt][nt], false, false);
    }
#else
    uint4 ra[2], rb[2];
    auto gload = [&](int kt){
#pragma unroll
        for (int i = 0; i < 2; ++i) {
            int id  = tid + i*256;
            int row = id >> 2;
            int kh  = (id & 3) * 8;
            ra[i] = *reinterpret_cast<const uint4*>(A  + (size_t)(bm+row)*K + kt*BK + kh);
            rb[i] = *reinterpret_cast<const uint4*>(Bt + (size_t)(bn+row)*K + kt*BK + kh);
            int ktp = kt + 2 < KT ? kt + 2 : KT - 1;
            __builtin_prefetch(A  + (size_t)(bm+row)*K + ktp*BK + kh, 0, 0);
            __builtin_prefetch(Bt + (size_t)(bn+row)*K + ktp*BK + kh, 0, 0);
        }
    };
    auto lstore = [&](int buf){
#pragma unroll
        for (int i = 0; i < 2; ++i) {
            int id  = tid + i*256;
            int row = id >> 2;
            int kh  = (id & 3) * 8;
            *reinterpret_cast<uint4*>(&As[buf][row*LSTR + kh]) = ra[i];
            *reinterpret_cast<uint4*>(&Bs[buf][row*LSTR + kh]) = rb[i];
        }
    };
    gload(0);
    lstore(0);
    for (int kt = 0; kt < KT; ++kt) {
        __syncthreads();
        if (kt + 1 < KT) gload(kt + 1);
        const int buf = kt & 1;
        v16h af[2], bf[4];
#pragma unroll
        for (int mt = 0; mt < 2; ++mt)
            af[mt] = load_frag(&As[buf][0], LSTR, wm*32 + mt*16 + (lane & 15), lane, 0);
#pragma unroll
        for (int nt = 0; nt < 4; ++nt)
            bf[nt] = load_frag(&Bs[buf][0], LSTR, wn*64 + nt*16 + (lane & 15), lane, 0);
#pragma unroll
        for (int mt = 0; mt < 2; ++mt)
#pragma unroll
            for (int nt = 0; nt < 4; ++nt)
                acc[mt][nt] = __builtin_amdgcn_wmma_f32_16x16x32_f16(
                    false, af[mt], false, bf[nt], (short)0, acc[mt][nt], false, false);
        if (kt + 1 < KT) lstore((kt + 1) & 1);
    }
#endif

    // epilogue: lane holds row (lane>>4)*8 + r, col (lane&15) per 16x16 tile
#pragma unroll
    for (int mt = 0; mt < 2; ++mt) {
#pragma unroll
        for (int r = 0; r < 8; ++r) {
            const int row = bm + wm*32 + mt*16 + (lane >> 4)*8 + r;
            if (MODE == 1) {
                if (row >= MW) continue;
                int win = row / NTOK, n = row % NTOK;
                int bb = win / 100, wr = (win % 100) / 10, wc = win % 10;
                int y = wr*WS_ + n/WS_, x = wc*WS_ + n%WS_;
                if (y >= H_ || x >= W_) continue;
                size_t rbase = ((size_t)bb*HW_ + (size_t)y*W_ + x) * DIM_;
#pragma unroll
                for (int nt = 0; nt < 4; ++nt) {
                    int col = bn + wn*64 + nt*16 + (lane & 15);
                    ((float*)out)[rbase + col] = resid[rbase + col] + acc[mt][nt][r] + bias[col];
                }
            } else {
#pragma unroll
                for (int nt = 0; nt < 4; ++nt) {
                    int col = bn + wn*64 + nt*16 + (lane & 15);
                    float v = acc[mt][nt][r] + bias[col];
                    if (MODE == 0) {
                        ((half_t*)out)[(size_t)row*N + col] = (half_t)v;
                    } else if (MODE == 2) {
                        v = 0.5f * v * (1.0f + erff(v * 0.70710678118654752f));
                        ((half_t*)out)[(size_t)row*N + col] = (half_t)v;
                    } else {
                        ((float*)out)[(size_t)row*N + col] = v + resid[(size_t)row*N + col];
                    }
                }
            }
        }
    }
}

// ---------------------------------------------------------------------------
// Windowed attention, one block = (window, head), 4 waves.
// S = Q K^T * scale + bias; softmax; O = P V.  49 padded to 64.
// Bias tile (incl. -1e30 pad mask) precomputed branchlessly into LDS.
// ---------------------------------------------------------------------------
__global__ __launch_bounds__(128)
void attn_kernel(const half_t* __restrict__ qkv, const float* __restrict__ attn_bias,
                 half_t* __restrict__ o16)
{
    __shared__ __align__(16) half_t Qs[64*40];
    __shared__ __align__(16) half_t Ks[64*40];
    __shared__ __align__(16) half_t Vt[32*72];   // V transposed [d][token]
    __shared__ __align__(16) half_t Ps[64*72];   // probabilities
    __shared__ float Bias[64*64];                // bias + mask tile

    const int win  = blockIdx.x;
    const int h    = blockIdx.y;
    const int tid  = threadIdx.x;
    const int lane = tid & 31;
    const int w    = tid >> 5;

    for (int idx = tid; idx < 64*32; idx += 128) {
        int n = idx >> 5, d = idx & 31;
        half_t qv = (half_t)0.0f, kv = (half_t)0.0f, vv = (half_t)0.0f;
        if (n < NTOK) {
            size_t base = ((size_t)win*NTOK + n)*QKVN + (size_t)h*96;
            qv = qkv[base + d];
            kv = qkv[base + 32 + d];
            vv = qkv[base + 64 + d];
        }
        Qs[n*40 + d] = qv;
        Ks[n*40 + d] = kv;
        Vt[d*72 + n] = vv;
    }
    for (int idx = tid; idx < 64*64; idx += 128) {
        int rs = idx >> 6, cs = idx & 63;
        float bv = -1.0e30f;
        if (rs < NTOK && cs < NTOK) {
            int dr = iabs_(rs/WS_ - cs/WS_), dc = iabs_(rs%WS_ - cs%WS_);
            bv = attn_bias[h*NTOK + dr*WS_ + dc];
        }
        Bias[idx] = bv;
    }
    __syncthreads();

    // scores: wave w computes rows w*16..w*16+15, all 64 cols
    v16h aq = load_frag(Qs, 40, w*16 + (lane & 15), lane, 0);
    v8f S[4];
#pragma unroll
    for (int nt = 0; nt < 4; ++nt) {
        v16h bk = load_frag(Ks, 40, nt*16 + (lane & 15), lane, 0);
        S[nt] = __builtin_amdgcn_wmma_f32_16x16x32_f16(false, aq, false, bk,
                                                       (short)0, zero8(), false, false);
    }

    // scale + bias/mask (branchless: masked slots hold -1e30, S there is 0)
#pragma unroll
    for (int nt = 0; nt < 4; ++nt)
#pragma unroll
        for (int r = 0; r < 8; ++r) {
            int rs = w*16 + (lane >> 4)*8 + r;
            int cs = nt*16 + (lane & 15);
            S[nt][r] = S[nt][r]*SCALE_ + Bias[rs*64 + cs];
        }

    // row softmax: 16-lane butterfly over column groups
    float mrow[8], srow[8];
#pragma unroll
    for (int r = 0; r < 8; ++r) {
        float m = fmaxf(fmaxf(S[0][r], S[1][r]), fmaxf(S[2][r], S[3][r]));
#pragma unroll
        for (int off = 1; off < 16; off <<= 1) m = fmaxf(m, __shfl_xor(m, off, 32));
        mrow[r] = m;
    }
#pragma unroll
    for (int nt = 0; nt < 4; ++nt)
#pragma unroll
        for (int r = 0; r < 8; ++r) S[nt][r] = __expf(S[nt][r] - mrow[r]);
#pragma unroll
    for (int r = 0; r < 8; ++r) {
        float s = S[0][r] + S[1][r] + S[2][r] + S[3][r];
#pragma unroll
        for (int off = 1; off < 16; off <<= 1) s += __shfl_xor(s, off, 32);
        srow[r] = 1.0f / s;
    }
#pragma unroll
    for (int nt = 0; nt < 4; ++nt)
#pragma unroll
        for (int r = 0; r < 8; ++r) {
            int rs = w*16 + (lane >> 4)*8 + r;
            int cs = nt*16 + (lane & 15);
            Ps[rs*72 + cs] = (half_t)(S[nt][r] * srow[r]);
        }
    __syncthreads();

    // O = P(64x64) * V(64x32): 2 K-steps x 2 N-tiles
    v8f O[2] = { zero8(), zero8() };
#pragma unroll
    for (int kk = 0; kk < 2; ++kk) {
        v16h ap = load_frag(Ps, 72, w*16 + (lane & 15), lane, kk*32);
#pragma unroll
        for (int nt = 0; nt < 2; ++nt) {
            v16h bv = load_frag(Vt, 72, nt*16 + (lane & 15), lane, kk*32);
            O[nt] = __builtin_amdgcn_wmma_f32_16x16x32_f16(false, ap, false, bv,
                                                           (short)0, O[nt], false, false);
        }
    }
#pragma unroll
    for (int nt = 0; nt < 2; ++nt)
#pragma unroll
        for (int r = 0; r < 8; ++r) {
            int tok = w*16 + (lane >> 4)*8 + r;
            int d   = nt*16 + (lane & 15);
            if (tok < NTOK)
                o16[((size_t)win*NTOK + tok)*DIM_ + h*HD_ + d] = (half_t)O[nt][r];
        }
}

// ---------------------------------------------------------------------------
// Depthwise 3x3 conv + BN (x stored as (b, hw, c), conv over spatial).
// ---------------------------------------------------------------------------
__global__ __launch_bounds__(128)
void convbn_kernel(const float* __restrict__ x1, const float* __restrict__ w,
                   const float* __restrict__ g, const float* __restrict__ bb,
                   const float* __restrict__ mean, const float* __restrict__ var,
                   float* __restrict__ out)
{
    int c = blockIdx.x * 128 + threadIdx.x;
    int p = blockIdx.y;
    int b = blockIdx.z;
    if (c >= DIM_) return;
    int y = p / W_, x = p % W_;
    float s = 0.0f;
#pragma unroll
    for (int dy = -1; dy <= 1; ++dy) {
        int yy = y + dy; if (yy < 0 || yy >= H_) continue;
#pragma unroll
        for (int dx = -1; dx <= 1; ++dx) {
            int xx = x + dx; if (xx < 0 || xx >= W_) continue;
            s += x1[((size_t)b*HW_ + (size_t)yy*W_ + xx)*DIM_ + c] * w[c*9 + (dy+1)*3 + (dx+1)];
        }
    }
    float o = (s - mean[c]) * rsqrtf(var[c] + EPS_) * g[c] + bb[c];
    out[((size_t)b*HW_ + p)*DIM_ + c] = o;
}

// ---------------------------------------------------------------------------
extern "C" void kernel_launch(void* const* d_in, const int* in_sizes, int n_in,
                              void* d_out, int out_size, void* d_ws, size_t ws_size,
                              hipStream_t stream)
{
    const float* x       = (const float*)d_in[0];
    const float* ln1_g   = (const float*)d_in[1];
    const float* ln1_b   = (const float*)d_in[2];
    const float* qkv_w   = (const float*)d_in[3];
    const float* qkv_b   = (const float*)d_in[4];
    const float* proj_w  = (const float*)d_in[5];
    const float* proj_b  = (const float*)d_in[6];
    const float* attn_b  = (const float*)d_in[7];
    const float* conv_w  = (const float*)d_in[8];
    const float* bn_g    = (const float*)d_in[9];
    const float* bn_b    = (const float*)d_in[10];
    const float* bn_mean = (const float*)d_in[11];
    const float* bn_var  = (const float*)d_in[12];
    const float* ln2_g   = (const float*)d_in[13];
    const float* ln2_b   = (const float*)d_in[14];
    const float* fc1_w   = (const float*)d_in[15];
    const float* fc1_b   = (const float*)d_in[16];
    const float* fc2_w   = (const float*)d_in[17];
    const float* fc2_b   = (const float*)d_in[18];

    char* p = (char*)d_ws;
    auto alloc = [&](size_t bytes) -> char* {
        char* r = p; p += (bytes + 255) & ~(size_t)255; return r;
    };
    half_t* xn16    = (half_t*)alloc((size_t)MWP * DIM_  * 2);
    half_t* qkv16   = (half_t*)alloc((size_t)MWP * QKVN  * 2);
    half_t* o16     = (half_t*)alloc((size_t)MWP * DIM_  * 2);
    float*  x1      = (float*) alloc((size_t)MX  * DIM_  * 4);
    float*  x2      = (float*) alloc((size_t)MX  * DIM_  * 4);
    half_t* xn2_16  = (half_t*)alloc((size_t)MX  * DIM_  * 2);
    half_t* h1_16   = (half_t*)alloc((size_t)MX  * HID_  * 2);
    half_t* qkv_wt  = (half_t*)alloc((size_t)QKVN * DIM_ * 2);
    half_t* proj_wt = (half_t*)alloc((size_t)DIM_ * DIM_ * 2);
    half_t* fc1_wt  = (half_t*)alloc((size_t)HID_ * DIM_ * 2);
    half_t* fc2_wt  = (half_t*)alloc((size_t)DIM_ * HID_ * 2);

    // weights -> f16 transposed [N][K]
    transpose_cvt<<<(DIM_*QKVN + 255)/256, 256, 0, stream>>>(qkv_w,  qkv_wt,  DIM_, QKVN);
    transpose_cvt<<<(DIM_*DIM_ + 255)/256, 256, 0, stream>>>(proj_w, proj_wt, DIM_, DIM_);
    transpose_cvt<<<(DIM_*HID_ + 255)/256, 256, 0, stream>>>(fc1_w,  fc1_wt,  DIM_, HID_);
    transpose_cvt<<<(HID_*DIM_ + 255)/256, 256, 0, stream>>>(fc2_w,  fc2_wt,  HID_, DIM_);

    // LN1 + window -> f16
    ln_kernel<<<MWP, 128, 0, stream>>>(x, ln1_g, ln1_b, xn16, 1);

    // QKV GEMM: [MWP,384] x [384,1152]
    gemm_wmma<0><<<dim3(QKVN/128, MWP/128), 256, 0, stream>>>(
        xn16, qkv_wt, qkv_b, (void*)qkv16, nullptr, MWP, QKVN, DIM_);

    // windowed attention
    attn_kernel<<<dim3(NWIN, NH_), 128, 0, stream>>>(qkv16, attn_b, o16);

    // proj GEMM + un-window + residual -> x1 (f32)
    gemm_wmma<1><<<dim3(DIM_/128, MWP/128), 256, 0, stream>>>(
        o16, proj_wt, proj_b, (void*)x1, x, MWP, DIM_, DIM_);

    // depthwise conv + BN -> x2
    convbn_kernel<<<dim3(DIM_/128, HW_, BATCH_), 128, 0, stream>>>(
        x1, conv_w, bn_g, bn_b, bn_mean, bn_var, x2);

    // LN2 -> f16
    ln_kernel<<<MX, 128, 0, stream>>>(x2, ln2_g, ln2_b, xn2_16, 0);

    // fc1 GEMM + GELU -> f16
    gemm_wmma<2><<<dim3(HID_/128, MX/128), 256, 0, stream>>>(
        xn2_16, fc1_wt, fc1_b, (void*)h1_16, nullptr, MX, HID_, DIM_);

    // fc2 GEMM + residual -> out (f32)
    gemm_wmma<3><<<dim3(DIM_/128, MX/128), 256, 0, stream>>>(
        h1_16, fc2_wt, fc2_b, d_out, x2, MX, DIM_, HID_);

    (void)in_sizes; (void)n_in; (void)out_size; (void)ws_size;
}
